// MoE_34935263985898
// MI455X (gfx1250) — matrix operand order repaired and checked by reading, not compiled, not run on hardware
//
#include <hip/hip_runtime.h>
#include <math.h>

typedef __attribute__((ext_vector_type(16))) __bf16 v16bf;
typedef __attribute__((ext_vector_type(2)))  __bf16 v2bf;
typedef __attribute__((ext_vector_type(8)))  float  v8f;

#define T_TOK 4096
#define HS 1024
#define FFN 4096
#define NEXP 8
#define CAP 1024

#define BM 128
#define BN 128
#define BKK 32
#define LDK 40   /* padded K stride in halfs: 80B rows -> 16B aligned, conflict-friendly */

/* let the compiler lower fptrunc -> native bf16 cvt (RNE) */
__device__ __forceinline__ unsigned short f32_to_bf16(float f) {
  __bf16 b = (__bf16)f;
  return __builtin_bit_cast(unsigned short, b);
}
__device__ __forceinline__ unsigned pack_bf16x2(float lo, float hi) {
  v2bf p = {(__bf16)lo, (__bf16)hi};
  return __builtin_bit_cast(unsigned, p);
}
/* branch-free tanh-gelu using hw transcendentals (v_exp_f32 / v_rcp_f32) */
__device__ __forceinline__ float gelu_tanh(float x) {
  const float c0 = 0.7978845608028654f;
  float z = c0 * (x + 0.044715f * x * x * x);
  float az = fabsf(z);
  float t = __builtin_amdgcn_exp2f(-2.8853900817779268f * az); /* exp(-2|z|) in (0,1] */
  float r = (1.f - t) * __builtin_amdgcn_rcpf(1.f + t);        /* tanh(|z|) */
  float th = copysignf(r, z);
  return 0.5f * x * (1.f + th);
}

union FragBF { uint4 q[2]; v16bf v; };

/* pack 16 fp32 -> 16 bf16, vectorized row store (K-contiguous) */
__device__ __forceinline__ void cvt_store_row16(unsigned short* dst, const float4 f[4]) {
  uint4 q0, q1;
  q0.x = pack_bf16x2(f[0].x, f[0].y); q0.y = pack_bf16x2(f[0].z, f[0].w);
  q0.z = pack_bf16x2(f[1].x, f[1].y); q0.w = pack_bf16x2(f[1].z, f[1].w);
  q1.x = pack_bf16x2(f[2].x, f[2].y); q1.y = pack_bf16x2(f[2].z, f[2].w);
  q1.z = pack_bf16x2(f[3].x, f[3].y); q1.w = pack_bf16x2(f[3].z, f[3].w);
  *(uint4*)dst = q0;
  *(uint4*)(dst + 8) = q1;
}
/* pack 16 fp32 -> bf16, transposed scatter store into [n][k] tile */
__device__ __forceinline__ void cvt_store_colT(unsigned short* Bsm, int bn, int bk,
                                               const float4 g[4]) {
  float v[16] = {g[0].x, g[0].y, g[0].z, g[0].w, g[1].x, g[1].y, g[1].z, g[1].w,
                 g[2].x, g[2].y, g[2].z, g[2].w, g[3].x, g[3].y, g[3].z, g[3].w};
#pragma unroll
  for (int j = 0; j < 16; j++) Bsm[(bn + j) * LDK + bk] = f32_to_bf16(v[j]);
}

/* ---------------- Kernel 1: router (softmax + top-2) ---------------- */
__global__ __launch_bounds__(256) void moe_router(const float* __restrict__ xf,
                                                  const float* __restrict__ rw,
                                                  int* __restrict__ te,
                                                  float* __restrict__ ew) {
  const int lane = threadIdx.x & 31;
  const int wid  = threadIdx.x >> 5;
  const int t = blockIdx.x * 8 + wid;
  float acc[NEXP];
#pragma unroll
  for (int e = 0; e < NEXP; e++) acc[e] = 0.f;
  const float4* rw4 = (const float4*)rw;
  for (int h = lane; h < HS; h += 32) {
    float xv = xf[(size_t)t * HS + h];
    float4 a = rw4[h * 2 + 0];
    float4 b = rw4[h * 2 + 1];
    acc[0] += xv * a.x; acc[1] += xv * a.y; acc[2] += xv * a.z; acc[3] += xv * a.w;
    acc[4] += xv * b.x; acc[5] += xv * b.y; acc[6] += xv * b.z; acc[7] += xv * b.w;
  }
#pragma unroll
  for (int e = 0; e < NEXP; e++)
    for (int off = 16; off >= 1; off >>= 1) acc[e] += __shfl_xor(acc[e], off, 32);
  if (lane == 0) {
    float m = acc[0];
#pragma unroll
    for (int e = 1; e < NEXP; e++) m = fmaxf(m, acc[e]);
    float p[NEXP];
    float s = 0.f;
#pragma unroll
    for (int e = 0; e < NEXP; e++) {
      p[e] = __builtin_amdgcn_exp2f((acc[e] - m) * 1.4426950408889634f);
      s += p[e];
    }
    float inv = 1.f / s;
#pragma unroll
    for (int e = 0; e < NEXP; e++) p[e] *= inv;
    int e1 = 0;
#pragma unroll
    for (int e = 1; e < NEXP; e++) if (p[e] > p[e1]) e1 = e;
    int e2 = (e1 == 0) ? 1 : 0;
#pragma unroll
    for (int e = 0; e < NEXP; e++) if (e != e1 && p[e] > p[e2]) e2 = e;
    te[t * 2 + 0] = e1; te[t * 2 + 1] = e2;
    ew[t * 2 + 0] = p[e1]; ew[t * 2 + 1] = p[e2];
  }
}

/* ------- Kernel 2: stable binning (histogram + prefix, no atomics) ------- */
__global__ __launch_bounds__(256) void moe_binning(const int* __restrict__ te,
                                                   int* __restrict__ slot_pos,
                                                   int* __restrict__ row_token) {
  __shared__ int cnt[256 * NEXP];
  const int t = threadIdx.x;
  for (int i = 0; i < 32; i++) row_token[t * 32 + i] = -1;  /* E*CAP == 8192 */
  int lc[NEXP];
#pragma unroll
  for (int e = 0; e < NEXP; e++) lc[e] = 0;
  const int base = t * 32;
  for (int i = 0; i < 32; i++) lc[te[base + i]]++;
#pragma unroll
  for (int e = 0; e < NEXP; e++) cnt[t * NEXP + e] = lc[e];
  __threadfence();
  __syncthreads();
  int pre[NEXP];
#pragma unroll
  for (int e = 0; e < NEXP; e++) pre[e] = 0;
  for (int tt = 0; tt < t; tt++)
#pragma unroll
    for (int e = 0; e < NEXP; e++) pre[e] += cnt[tt * NEXP + e];
  for (int i = 0; i < 32; i++) {
    int s = base + i;
    int e = te[s];
    int r = pre[e]++;
    if (r < CAP) { slot_pos[s] = r; row_token[e * CAP + r] = s >> 1; }
    else slot_pos[s] = -1;
  }
}

/* ---- Kernel 3: GEMM1  h = gelu(gather(x) @ w1); bf16 WMMA, reg-prefetch pipeline ---- */
__global__ __launch_bounds__(256) void moe_gemm1(const float* __restrict__ xf,
                                                 const float* __restrict__ w1,
                                                 const int* __restrict__ row_token,
                                                 unsigned short* __restrict__ hbuf) {
  __shared__ unsigned short As[BM * LDK];
  __shared__ unsigned short Bs[BN * LDK];
  const int e = blockIdx.z;
  const int nBase = blockIdx.x * BN;
  const int cBase = blockIdx.y * BM;
  const int tid = threadIdx.x;
  const int lane = tid & 31;
  const int wid = tid >> 5;
  const int wm = wid >> 2;
  const int wn = wid & 3;

  const int ar = tid >> 1;
  const int ak = (tid & 1) * 16;
  const int tok = row_token[e * CAP + cBase + ar];
  const bool avalid = (tok >= 0);
  const float* __restrict__ arow = xf + (size_t)(avalid ? tok : 0) * HS;

  const int bk = tid >> 3;
  const int bn = (tid & 7) * 16;
  const float* __restrict__ w1e = w1 + (size_t)e * HS * FFN;

  v8f acc[4][2];
  const v8f vz = {0.f, 0.f, 0.f, 0.f, 0.f, 0.f, 0.f, 0.f};
#pragma unroll
  for (int mi = 0; mi < 4; mi++)
#pragma unroll
    for (int ni = 0; ni < 2; ni++) acc[mi][ni] = vz;

  float4 fa[4] = {{0,0,0,0},{0,0,0,0},{0,0,0,0},{0,0,0,0}};
  float4 fb[4];
  if (avalid) {
    const float4* p = (const float4*)(arow + ak);
    fa[0] = p[0]; fa[1] = p[1]; fa[2] = p[2]; fa[3] = p[3];
  }
  {
    const float4* p4 = (const float4*)(w1e + (size_t)bk * FFN + nBase + bn);
    fb[0] = p4[0]; fb[1] = p4[1]; fb[2] = p4[2]; fb[3] = p4[3];
  }

  for (int k0 = 0; k0 < HS; k0 += BKK) {
    cvt_store_row16(&As[ar * LDK + ak], fa);
    cvt_store_colT(Bs, bn, bk, fb);
    /* register-prefetch next K-slab while this slab is consumed by WMMA */
    if (k0 + BKK < HS) {
      if (avalid) {
        const float4* p = (const float4*)(arow + k0 + BKK + ak);
        fa[0] = p[0]; fa[1] = p[1]; fa[2] = p[2]; fa[3] = p[3];
      }
      const float4* p4 = (const float4*)(w1e + (size_t)(k0 + BKK + bk) * FFN + nBase + bn);
      fb[0] = p4[0]; fb[1] = p4[1]; fb[2] = p4[2]; fb[3] = p4[3];
    }
    __syncthreads();

    FragBF af[4];
    const int koffA = (lane >= 16) ? 8 : 0;
#pragma unroll
    for (int mi = 0; mi < 4; mi++) {
      int row = wm * 64 + mi * 16 + (lane & 15);
      af[mi].q[0] = *(const uint4*)&As[row * LDK + koffA];
      af[mi].q[1] = *(const uint4*)&As[row * LDK + koffA + 16];
    }
    FragBF bfg[2];
    const int kb = (lane >= 16) ? 8 : 0;
#pragma unroll
    for (int ni = 0; ni < 2; ni++) {
      int col = wn * 32 + ni * 16 + (lane & 15);
      bfg[ni].q[0] = *(const uint4*)&Bs[col * LDK + kb];
      bfg[ni].q[1] = *(const uint4*)&Bs[col * LDK + kb + 16];
    }
#pragma unroll
    for (int mi = 0; mi < 4; mi++)
#pragma unroll
      for (int ni = 0; ni < 2; ni++)
        acc[mi][ni] = __builtin_amdgcn_wmma_f32_16x16x32_bf16(
            false, af[mi].v, false, bfg[ni].v, (short)0, acc[mi][ni], false, false);
    __syncthreads();
  }

  const int rbase = (lane >= 16) ? 8 : 0;
  const int coln = lane & 15;
#pragma unroll
  for (int mi = 0; mi < 4; mi++)
#pragma unroll
    for (int ni = 0; ni < 2; ni++) {
      int col = nBase + wn * 32 + ni * 16 + coln;
#pragma unroll
      for (int r = 0; r < 8; r++) {
        int row = cBase + wm * 64 + mi * 16 + rbase + r;
        float v = gelu_tanh(acc[mi][ni][r]);
        hbuf[((size_t)e * CAP + row) * FFN + col] = f32_to_bf16(v);
      }
    }
}

/* ---- Kernel 4: GEMM2  y = h @ w2; A via double-buffered async global->LDS ---- */
__global__ __launch_bounds__(256) void moe_gemm2(const unsigned short* __restrict__ hbuf,
                                                 const float* __restrict__ w2,
                                                 float* __restrict__ ybuf) {
  __shared__ unsigned short As[2][BM * LDK];
  __shared__ unsigned short Bs[BN * LDK];
  const int e = blockIdx.z;
  const int nBase = blockIdx.x * BN;  /* over HS */
  const int cBase = blockIdx.y * BM;
  const int tid = threadIdx.x;
  const int lane = tid & 31;
  const int wid = tid >> 5;
  const int wm = wid >> 2;
  const int wn = wid & 3;

  const int ar = tid >> 1;
  const int ak = (tid & 1) * 16;
  const unsigned long long aBase =
      (unsigned long long)(const void*)(hbuf + ((size_t)e * CAP + cBase) * FFN);
  const unsigned aoff = (unsigned)((ar * LDK + ak) * 2);
  const unsigned dstA0 = (unsigned)(size_t)(void*)&As[0][0] + aoff;
  const unsigned dstA1 = (unsigned)(size_t)(void*)&As[1][0] + aoff;

  const int bk = tid >> 3;
  const int bn = (tid & 7) * 16;
  const float* __restrict__ w2e = w2 + (size_t)e * FFN * HS;

  v8f acc[4][2];
  const v8f vz = {0.f, 0.f, 0.f, 0.f, 0.f, 0.f, 0.f, 0.f};
#pragma unroll
  for (int mi = 0; mi < 4; mi++)
#pragma unroll
    for (int ni = 0; ni < 2; ni++) acc[mi][ni] = vz;

  /* prologue: async A slab 0 -> buffer 0; B slab 0 -> registers */
  {
    unsigned voff = (unsigned)(((size_t)ar * FFN + ak) * 2);
    asm volatile("global_load_async_to_lds_b128 %0, %1, %2"
                 :: "v"(dstA0), "v"(voff), "s"(aBase) : "memory");
    asm volatile("global_load_async_to_lds_b128 %0, %1, %2 offset:16"
                 :: "v"(dstA0), "v"(voff), "s"(aBase) : "memory");
  }
  float4 fb[4];
  {
    const float4* p4 = (const float4*)(w2e + (size_t)bk * HS + nBase + bn);
    fb[0] = p4[0]; fb[1] = p4[1]; fb[2] = p4[2]; fb[3] = p4[3];
  }

  int cur = 0;
  for (int k0 = 0; k0 < FFN; k0 += BKK, cur ^= 1) {
    cvt_store_colT(Bs, bn, bk, fb);
    if (k0 + BKK < FFN) {
      /* issue next A slab into the other buffer; prefetch next B slab to regs */
      unsigned voff = (unsigned)(((size_t)ar * FFN + k0 + BKK + ak) * 2);
      unsigned dstN = cur ? dstA0 : dstA1;
      asm volatile("global_load_async_to_lds_b128 %0, %1, %2"
                   :: "v"(dstN), "v"(voff), "s"(aBase) : "memory");
      asm volatile("global_load_async_to_lds_b128 %0, %1, %2 offset:16"
                   :: "v"(dstN), "v"(voff), "s"(aBase) : "memory");
      const float4* p4 = (const float4*)(w2e + (size_t)(k0 + BKK + bk) * HS + nBase + bn);
      fb[0] = p4[0]; fb[1] = p4[1]; fb[2] = p4[2]; fb[3] = p4[3];
      /* current slab done when only next slab's pair is outstanding */
      asm volatile("s_wait_asynccnt 2" ::: "memory");
    } else {
      asm volatile("s_wait_asynccnt 0" ::: "memory");
    }
    __syncthreads();

    const unsigned short* Acur = &As[cur][0];
    FragBF af[4];
    const int koffA = (lane >= 16) ? 8 : 0;
#pragma unroll
    for (int mi = 0; mi < 4; mi++) {
      int row = wm * 64 + mi * 16 + (lane & 15);
      af[mi].q[0] = *(const uint4*)&Acur[row * LDK + koffA];
      af[mi].q[1] = *(const uint4*)&Acur[row * LDK + koffA + 16];
    }
    FragBF bfg[2];
    const int kb = (lane >= 16) ? 8 : 0;
#pragma unroll
    for (int ni = 0; ni < 2; ni++) {
      int col = wn * 32 + ni * 16 + (lane & 15);
      bfg[ni].q[0] = *(const uint4*)&Bs[col * LDK + kb];
      bfg[ni].q[1] = *(const uint4*)&Bs[col * LDK + kb + 16];
    }
#pragma unroll
    for (int mi = 0; mi < 4; mi++)
#pragma unroll
      for (int ni = 0; ni < 2; ni++)
        acc[mi][ni] = __builtin_amdgcn_wmma_f32_16x16x32_bf16(
            false, af[mi].v, false, bfg[ni].v, (short)0, acc[mi][ni], false, false);
    __syncthreads();
  }

  const int rbase = (lane >= 16) ? 8 : 0;
  const int coln = lane & 15;
#pragma unroll
  for (int mi = 0; mi < 4; mi++)
#pragma unroll
    for (int ni = 0; ni < 2; ni++) {
      int col = nBase + wn * 32 + ni * 16 + coln;
#pragma unroll
      for (int r = 0; r < 8; r++) {
        int row = cBase + wm * 64 + mi * 16 + rbase + r;
        ybuf[((size_t)e * CAP + row) * HS + col] = acc[mi][ni][r];
      }
    }
}

/* ---- Kernel 5: weighted combine (gather form of scatter, deterministic) ---- */
__global__ __launch_bounds__(256) void moe_combine(const float* __restrict__ ybuf,
                                                   const int* __restrict__ te,
                                                   const float* __restrict__ ew,
                                                   const int* __restrict__ slot_pos,
                                                   float* __restrict__ out) {
  int idx = blockIdx.x * 256 + threadIdx.x;  /* float4 index */
  int t = idx >> 8;                          /* HS/4 = 256 float4 per token */
  int c4 = idx & 255;
  float4 acc = {0.f, 0.f, 0.f, 0.f};
#pragma unroll
  for (int k = 0; k < 2; k++) {
    int s = t * 2 + k;
    int pos = slot_pos[s];
    if (pos >= 0) {
      int e = te[s];
      float w = ew[s];
      const float4* yp = (const float4*)(ybuf + (size_t)(e * CAP + pos) * HS);
      float4 v = yp[c4];
      acc.x += w * v.x; acc.y += w * v.y; acc.z += w * v.z; acc.w += w * v.w;
    }
  }
  ((float4*)out)[idx] = acc;
}

/* ---- Kernel 6: copy bias (second tuple output) ---- */
__global__ __launch_bounds__(256) void moe_bias(const float* __restrict__ bias,
                                                float* __restrict__ dst) {
  ((float4*)dst)[threadIdx.x] = ((const float4*)bias)[threadIdx.x];
}

extern "C" void kernel_launch(void* const* d_in, const int* in_sizes, int n_in,
                              void* d_out, int out_size, void* d_ws, size_t ws_size,
                              hipStream_t stream) {
  const float* x    = (const float*)d_in[0];
  const float* rw   = (const float*)d_in[1];
  const float* w1   = (const float*)d_in[2];
  const float* w2   = (const float*)d_in[3];
  const float* bias = (const float*)d_in[4];
  float* out = (float*)d_out;

  char* ws = (char*)d_ws;
  int*   te        = (int*)(ws + 0);
  float* ew        = (float*)(ws + 32768);
  int*   slot_pos  = (int*)(ws + 65536);
  int*   row_token = (int*)(ws + 98304);
  unsigned short* hbuf = (unsigned short*)(ws + 131072);
  float* ybuf = (float*)(ws + 131072 + (size_t)NEXP * CAP * FFN * 2);

  moe_router<<<T_TOK / 8, 256, 0, stream>>>(x, rw, te, ew);
  moe_binning<<<1, 256, 0, stream>>>(te, slot_pos, row_token);
  moe_gemm1<<<dim3(FFN / BN, CAP / BM, NEXP), 256, 0, stream>>>(x, w1, row_token, hbuf);
  moe_gemm2<<<dim3(HS / BN, CAP / BM, NEXP), 256, 0, stream>>>(hbuf, w2, ybuf);
  moe_combine<<<(T_TOK * HS / 4) / 256, 256, 0, stream>>>(ybuf, te, ew, slot_pos, out);
  moe_bias<<<1, 256, 0, stream>>>(bias, out + (size_t)T_TOK * HS);
}
